// qQrohQXiAITQznFuShydRodggjOWexBZ_84980222919382
// MI455X (gfx1250) — matrix operand order repaired and checked by reading, not compiled, not run on hardware
//
#include <hip/hip_runtime.h>
#include <hip/hip_bf16.h>

typedef __attribute__((ext_vector_type(16))) __bf16 v16bf;
typedef __attribute__((ext_vector_type(8)))  float  v8f;
typedef unsigned int tdm_v4u __attribute__((ext_vector_type(4)));
typedef int          tdm_v8i __attribute__((ext_vector_type(8)));
typedef int          tdm_v4i __attribute__((ext_vector_type(4)));

#define HEADS    8
#define DIM_HEAD 64
#define SEQ      4096
#define BATCH    2
#define DMODEL   1024
#define INNER    512
#define NQKV     1536
#define FA_ITERS (SEQ / 32)

__device__ __forceinline__ __bf16 f2bf(float f) {
  unsigned u = __builtin_bit_cast(unsigned, f);
  unsigned r = u + 0x7FFFu + ((u >> 16) & 1u);   // round to nearest even
  unsigned short s = (unsigned short)(r >> 16);
  return __builtin_bit_cast(__bf16, s);
}

__device__ __forceinline__ v8f wmma_bf16(v16bf a, v16bf b, v8f c) {
  return __builtin_amdgcn_wmma_f32_16x16x32_bf16(false, a, false, b, (short)0, c,
                                                 false, false);
}

// 16-lane xor-butterfly reductions via DPP16 (stay in VALU, co-issue with WMMA)
template <int CTRL>
__device__ __forceinline__ float dpp_max(float v) {
  int s = __builtin_amdgcn_update_dpp(0, __builtin_bit_cast(int, v), CTRL, 0xF, 0xF, true);
  return fmaxf(v, __builtin_bit_cast(float, s));
}
template <int CTRL>
__device__ __forceinline__ float dpp_add(float v) {
  int s = __builtin_amdgcn_update_dpp(0, __builtin_bit_cast(int, v), CTRL, 0xF, 0xF, true);
  return v + __builtin_bit_cast(float, s);
}
__device__ __forceinline__ float rmax16(float v) {
  v = dpp_max<0xB1>(v);    // quad_perm(1,0,3,2)  : xor 1
  v = dpp_max<0x4E>(v);    // quad_perm(2,3,0,1)  : xor 2
  v = dpp_max<0x141>(v);   // row_half_mirror     : xor 4
  v = dpp_max<0x140>(v);   // row_mirror          : xor 8
  return v;
}
__device__ __forceinline__ float rsum16(float v) {
  v = dpp_add<0xB1>(v);
  v = dpp_add<0x4E>(v);
  v = dpp_add<0x141>(v);
  v = dpp_add<0x140>(v);
  return v;
}

// ---------------------------------------------------------------------------
// Tensor Data Mover: async-copy one contiguous 4KB (2048 x bf16) block into
// LDS.  1-D D#: group0 = {count=1, lds_addr, global_addr, type=2},
// group1 = {data_size=2B, tensor_dim0=2048, tile_dim0=2048}.  Tracked by
// TENSORcnt; caller pairs with s_wait_tensorcnt.
// ---------------------------------------------------------------------------
__device__ __forceinline__ void tdm_load_4kb(unsigned lds_off, const void* gptr) {
  unsigned long long ga = (unsigned long long)(uintptr_t)gptr;
  tdm_v4u g0 = { 1u,                                   // count=1, user mode
                 lds_off,                              // LDS byte address
                 (unsigned)ga,                         // global addr [31:0]
                 (unsigned)((ga >> 32) & 0x01FFFFFFull) | (2u << 30) }; // [56:32] | type=2
  const unsigned td0 = 2048u;                          // elements (2B each)
  tdm_v8i g1 = {
    (int)(1u << 16),                 // workgroup_mask=0, data_size=1 (2 bytes)
    (int)((td0 & 0xFFFFu) << 16),    // tensor_dim0[15:0] (bits 63:48)
    (int)((td0 >> 16) & 0xFFFFu),    // tensor_dim0[31:16]; tensor_dim1=0
    (int)(td0 << 16),                // tile_dim0 = 2048 (bits 127:112)
    0,                               // tile_dim1 = 0 (1-D), tile_dim2 = 0
    (int)td0,                        // tensor_dim0_stride lo
    0, 0
  };
  tdm_v4i z4 = {0, 0, 0, 0};
  tdm_v8i z8 = {0, 0, 0, 0, 0, 0, 0, 0};
  __builtin_amdgcn_tensor_load_to_lds(g0, g1, z4, z4, z8, 0);
}

// ---------------------------------------------------------------------------
// Kernel 0: fp32 -> bf16 weight packing.  Wqkvb is [1024][1536] (q|k|v cols).
// ---------------------------------------------------------------------------
__global__ __launch_bounds__(256) void pack_weights(
    const float* __restrict__ Wq, const float* __restrict__ Wk,
    const float* __restrict__ Wv, const float* __restrict__ Wout,
    __bf16* __restrict__ Wqkvb, __bf16* __restrict__ Woutb)
{
  int i = blockIdx.x * 256 + threadIdx.x;
  const int NQ = DMODEL * NQKV;
  if (i < NQ) {
    int k = i / NQKV, c = i % NQKV;
    float v = (c < INNER)      ? Wq[(size_t)k * INNER + c]
            : (c < 2 * INNER)  ? Wk[(size_t)k * INNER + (c - INNER)]
                               : Wv[(size_t)k * INNER + (c - 2 * INNER)];
    Wqkvb[i] = f2bf(v);
  } else {
    int j = i - NQ;
    if (j < INNER * DMODEL) Woutb[j] = f2bf(Wout[j]);
  }
}

// ---------------------------------------------------------------------------
// Tiled bf16 WMMA GEMM: C[128x128 block] = A[M,K] * B[K,N]
//   AF32: A is fp32, converted to bf16 while staging into LDS.
//   EPI 0: scatter bf16 result into Q/K/V  [b,h,n,dh]
//   EPI 1: fp32 result + bias -> outp [M,N]
// ---------------------------------------------------------------------------
template <bool AF32, int EPI>
__global__ __launch_bounds__(256) void gemm_k(
    const void* __restrict__ Aptr, const __bf16* __restrict__ B,
    int ldA, int N, int K,
    __bf16* __restrict__ qb, __bf16* __restrict__ kb, __bf16* __restrict__ vb,
    float* __restrict__ outp, const float* __restrict__ bias)
{
  __shared__ __align__(16) __bf16 sA[128 * 32];   // [row][k]
  __shared__ __align__(16) __bf16 sBt[128 * 32];  // transposed: [col][k]

  const int t    = threadIdx.x;
  const int lane = t & 31;
  const int wave = t >> 5;
  const int lo   = lane & 15;
  const int hi   = (lane & 16) ? 1 : 0;
  const int m0   = blockIdx.x * 128;
  const int n0   = blockIdx.y * 128;

  v8f acc[8];
  {
    v8f zz = {0.f, 0.f, 0.f, 0.f, 0.f, 0.f, 0.f, 0.f};
#pragma unroll
    for (int g = 0; g < 8; ++g) acc[g] = zz;
  }

  for (int k0 = 0; k0 < K; k0 += 32) {
    __syncthreads();
    if constexpr (AF32) {
      const float* Af = (const float*)Aptr;
#pragma unroll
      for (int u = t; u < 512; u += 256) {
        int row = u >> 2, c8 = (u & 3) * 8;
        const float* src = Af + (size_t)(m0 + row) * ldA + k0 + c8;
        float4 f0 = *(const float4*)src;
        float4 f1 = *(const float4*)(src + 4);
        __bf16* d = &sA[row * 32 + c8];
        d[0] = f2bf(f0.x); d[1] = f2bf(f0.y); d[2] = f2bf(f0.z); d[3] = f2bf(f0.w);
        d[4] = f2bf(f1.x); d[5] = f2bf(f1.y); d[6] = f2bf(f1.z); d[7] = f2bf(f1.w);
      }
    } else {
      const __bf16* Ab = (const __bf16*)Aptr;
#pragma unroll
      for (int u = t; u < 512; u += 256) {
        int row = u >> 2, c8 = (u & 3) * 8;
        *(uint4*)&sA[row * 32 + c8] =
            *(const uint4*)(Ab + (size_t)(m0 + row) * ldA + k0 + c8);
      }
    }
    // B tile 32x128, stored transposed so K runs contiguously per column.
#pragma unroll
    for (int u = t; u < 512; u += 256) {
      int row = u >> 4, c8 = (u & 15) * 8;
      uint4 vec = *(const uint4*)(B + (size_t)(k0 + row) * N + n0 + c8);
      __bf16 tmp[8];
      *(uint4*)tmp = vec;
#pragma unroll
      for (int i = 0; i < 8; ++i) sBt[(c8 + i) * 32 + row] = tmp[i];
    }
    // prefetch next k-tile into L2 while this one is consumed
    if (k0 + 32 < K) {
      if constexpr (AF32) {
        const float* Af = (const float*)Aptr;
        __builtin_prefetch(Af + (size_t)(m0 + (t >> 1)) * ldA + k0 + 32, 0, 1);
      } else {
        const __bf16* Ab = (const __bf16*)Aptr;
        __builtin_prefetch(Ab + (size_t)(m0 + (t >> 1)) * ldA + k0 + 32, 0, 1);
      }
      __builtin_prefetch(B + (size_t)(k0 + 32 + (t >> 3)) * N + n0 + (t & 7) * 16, 0, 1);
    }
    __syncthreads();

    v16bf a;
    {
      int arow = wave * 16 + lo;
      int aoff = hi * 16;                   // A layout: K in [off,off+16)
      ((uint4*)&a)[0] = *(uint4*)&sA[arow * 32 + aoff];
      ((uint4*)&a)[1] = *(uint4*)&sA[arow * 32 + aoff + 8];
    }
    const int koff = hi * 16;               // B layout: K in [koff, koff+16)
#pragma unroll
    for (int g = 0; g < 8; ++g) {
      v16bf bb;
      ((uint4*)&bb)[0] = *(uint4*)&sBt[(g * 16 + lo) * 32 + koff];
      ((uint4*)&bb)[1] = *(uint4*)&sBt[(g * 16 + lo) * 32 + koff + 8];
      acc[g] = wmma_bf16(a, bb, acc[g]);
    }
  }

  // epilogue: C layout -> lane lo = col, VGPR r = row r (+8 for hi lanes)
#pragma unroll
  for (int g = 0; g < 8; ++g) {
    int c = n0 + g * 16 + lo;
#pragma unroll
    for (int r = 0; r < 8; ++r) {
      int m   = m0 + wave * 16 + r + hi * 8;
      float v = acc[g][r];
      if constexpr (EPI == 0) {
        int which = c >> 9;          // 0:q 1:k 2:v
        int hh    = (c >> 6) & 7;
        int dd    = c & 63;
        int bi    = m >> 12;
        int nn    = m & 4095;
        __bf16* dst = (which == 0) ? qb : (which == 1) ? kb : vb;
        dst[(((size_t)(bi * HEADS + hh)) * SEQ + nn) * DIM_HEAD + dd] = f2bf(v);
      } else {
        outp[(size_t)m * N + c] = v + bias[c];
      }
    }
  }
}

// ---------------------------------------------------------------------------
// Flash attention: one block per (b*h, 128 q-rows).  8 waves x 16 rows.
// K/V tiles (32 keys x 64 dh, 4KB each) double-buffered in LDS via the
// Tensor Data Mover; wave 0 prefetches tile i+1 while everyone computes i.
// ---------------------------------------------------------------------------
__global__ __launch_bounds__(256) void flash_attn(
    const __bf16* __restrict__ Qb, const __bf16* __restrict__ Kb,
    const __bf16* __restrict__ Vb, __bf16* __restrict__ Ob)
{
  __shared__ __align__(16) __bf16 sK[2][32 * 64];  // [key][dh]
  __shared__ __align__(16) __bf16 sV[2][32 * 64];  // [key][dh] (linear)
  __shared__ __align__(16) __bf16 sP[8][16 * 32];  // per-wave P exchange

  const int t    = threadIdx.x;
  const int lane = t & 31;
  const int wave = t >> 5;
  const int lo   = lane & 15;
  const int hi   = (lane & 16) ? 1 : 0;
  const int bh   = blockIdx.x;
  const int bi   = bh >> 3, hh = bh & 7;
  const int row0 = blockIdx.y * 128;

  const __bf16* Qbh = Qb + (size_t)bh * SEQ * DIM_HEAD;
  const __bf16* Kbh = Kb + (size_t)bh * SEQ * DIM_HEAD;
  const __bf16* Vbh = Vb + (size_t)bh * SEQ * DIM_HEAD;

  // q fragments for this wave's 16 rows, held in registers all loop
  v16bf aq0, aq1;
  {
    int mrow = row0 + wave * 16 + lo;
    const __bf16* qp = Qbh + (size_t)mrow * DIM_HEAD;
    int aoff = hi * 16;
    ((uint4*)&aq0)[0] = *(const uint4*)(qp + aoff);
    ((uint4*)&aq0)[1] = *(const uint4*)(qp + aoff + 8);
    ((uint4*)&aq1)[0] = *(const uint4*)(qp + 32 + aoff);
    ((uint4*)&aq1)[1] = *(const uint4*)(qp + 32 + aoff + 8);
  }

  v8f zz = {0.f, 0.f, 0.f, 0.f, 0.f, 0.f, 0.f, 0.f};
  v8f acc[4];
#pragma unroll
  for (int g = 0; g < 4; ++g) acc[g] = zz;
  float mrun[8], lrun[8];
#pragma unroll
  for (int r = 0; r < 8; ++r) { mrun[r] = -1e30f; lrun[r] = 0.f; }

  const float SCALE = 0.125f;                 // 64^-0.5
  const float L2E   = 1.4426950408889634f;

  // prologue: kick off tile 0
  if (wave == 0) {
    tdm_load_4kb((unsigned)(uintptr_t)(const void*)&sK[0][0], Kbh);
    tdm_load_4kb((unsigned)(uintptr_t)(const void*)&sV[0][0], Vbh);
  }

  for (int it = 0; it < FA_ITERS; ++it) {
    const int buf = it & 1;
    __syncthreads();                 // everyone done with buf^1 from iter it-1
    if (wave == 0) {
      if (it + 1 < FA_ITERS) {
        const int jn = (it + 1) * 32;
        tdm_load_4kb((unsigned)(uintptr_t)(const void*)&sK[buf ^ 1][0],
                     Kbh + (size_t)jn * DIM_HEAD);
        tdm_load_4kb((unsigned)(uintptr_t)(const void*)&sV[buf ^ 1][0],
                     Vbh + (size_t)jn * DIM_HEAD);
        __builtin_amdgcn_s_wait_tensorcnt(2);   // tile `it` complete (in-order)
      } else {
        __builtin_amdgcn_s_wait_tensorcnt(0);
      }
    }
    __syncthreads();                 // tile `it` visible to all waves

    const __bf16* kp = &sK[buf][0];
    const __bf16* vp = &sV[buf][0];
    const int koff = hi * 16;

    v8f s0, s1;
    {   // S = q . k^T : two 16x16 key subtiles, dh split into two K=32 steps
      v16bf bk;
      ((uint4*)&bk)[0] = *(uint4*)&kp[lo * 64 + koff];
      ((uint4*)&bk)[1] = *(uint4*)&kp[lo * 64 + koff + 8];
      s0 = wmma_bf16(aq0, bk, zz);
      ((uint4*)&bk)[0] = *(uint4*)&kp[lo * 64 + 32 + koff];
      ((uint4*)&bk)[1] = *(uint4*)&kp[lo * 64 + 32 + koff + 8];
      s0 = wmma_bf16(aq1, bk, s0);
      ((uint4*)&bk)[0] = *(uint4*)&kp[(16 + lo) * 64 + koff];
      ((uint4*)&bk)[1] = *(uint4*)&kp[(16 + lo) * 64 + koff + 8];
      s1 = wmma_bf16(aq0, bk, zz);
      ((uint4*)&bk)[0] = *(uint4*)&kp[(16 + lo) * 64 + 32 + koff];
      ((uint4*)&bk)[1] = *(uint4*)&kp[(16 + lo) * 64 + 32 + koff + 8];
      s1 = wmma_bf16(aq1, bk, s1);
    }

    // online softmax; row r lives in element r across a 16-lane half-wave
#pragma unroll
    for (int r = 0; r < 8; ++r) {
      float a = s0[r] * SCALE;
      float b = s1[r] * SCALE;
      float tmax  = rmax16(fmaxf(a, b));
      float mnew  = fmaxf(mrun[r], tmax);
      float alpha = exp2f((mrun[r] - mnew) * L2E);
      float p0    = exp2f((a - mnew) * L2E);
      float p1    = exp2f((b - mnew) * L2E);
      float rs    = rsum16(p0 + p1);
      lrun[r] = lrun[r] * alpha + rs;
      mrun[r] = mnew;
#pragma unroll
      for (int g = 0; g < 4; ++g) acc[g][r] *= alpha;
      int prow = r + hi * 8;
      sP[wave][prow * 32 + lo]      = f2bf(p0);
      sP[wave][prow * 32 + 16 + lo] = f2bf(p1);
    }
    // wave-local LDS exchange: C-layout P -> A-fragment layout
    __asm__ volatile("s_wait_dscnt 0" ::: "memory");
    v16bf ap;
    {
      int aoff = hi * 16;
      ((uint4*)&ap)[0] = *(uint4*)&sP[wave][lo * 32 + aoff];
      ((uint4*)&ap)[1] = *(uint4*)&sP[wave][lo * 32 + aoff + 8];
    }
#pragma unroll
    for (int g = 0; g < 4; ++g) {   // O += P . V  (dh in 4 groups of 16)
      v16bf bv;
#pragma unroll
      for (int e = 0; e < 16; ++e)
        bv[e] = vp[(koff + e) * 64 + g * 16 + lo];
      acc[g] = wmma_bf16(ap, bv, acc[g]);
    }
  }

  float inv[8];
#pragma unroll
  for (int r = 0; r < 8; ++r) inv[r] = 1.0f / lrun[r];
#pragma unroll
  for (int g = 0; g < 4; ++g) {
#pragma unroll
    for (int r = 0; r < 8; ++r) {
      int nn  = row0 + wave * 16 + r + hi * 8;
      float o = acc[g][r] * inv[r];
      // Ob layout: [b*n][h*dh]  (ready for the output projection GEMM)
      Ob[((size_t)(bi * SEQ) + nn) * INNER + hh * DIM_HEAD + g * 16 + lo] = f2bf(o);
    }
  }
}

// ---------------------------------------------------------------------------
extern "C" void kernel_launch(void* const* d_in, const int* in_sizes, int n_in,
                              void* d_out, int out_size, void* d_ws, size_t ws_size,
                              hipStream_t stream) {
  (void)in_sizes; (void)n_in; (void)out_size; (void)ws_size;
  const float* x    = (const float*)d_in[0];
  const float* Wq   = (const float*)d_in[1];
  const float* Wk   = (const float*)d_in[2];
  const float* Wv   = (const float*)d_in[3];
  const float* Wout = (const float*)d_in[4];
  const float* bout = (const float*)d_in[5];
  float* out        = (float*)d_out;

  __bf16* p     = (__bf16*)d_ws;
  __bf16* Wqkvb = p; p += (size_t)DMODEL * NQKV;                    // 3 MB
  __bf16* Woutb = p; p += (size_t)INNER * DMODEL;                   // 1 MB
  __bf16* Qb    = p; p += (size_t)BATCH * HEADS * SEQ * DIM_HEAD;   // 8 MB
  __bf16* Kb    = p; p += (size_t)BATCH * HEADS * SEQ * DIM_HEAD;   // 8 MB
  __bf16* Vb    = p; p += (size_t)BATCH * HEADS * SEQ * DIM_HEAD;   // 8 MB
  __bf16* Ob    = p;                                                // 8 MB

  pack_weights<<<(DMODEL * NQKV + INNER * DMODEL) / 256, 256, 0, stream>>>(
      Wq, Wk, Wv, Wout, Wqkvb, Woutb);

  // QKV projection: [8192,1024] x [1024,1536]
  gemm_k<true, 0><<<dim3(64, 12), 256, 0, stream>>>(
      x, Wqkvb, DMODEL, NQKV, DMODEL, Qb, Kb, Vb, nullptr, nullptr);

  // attention: 16 (b,h) pairs x 32 row-blocks
  flash_attn<<<dim3(16, 32), 256, 0, stream>>>(Qb, Kb, Vb, Ob);

  // output projection: [8192,512] x [512,1024] + bias
  gemm_k<false, 1><<<dim3(64, 8), 256, 0, stream>>>(
      Ob, Woutb, INNER, DMODEL, INNER, nullptr, nullptr, nullptr, out, bout);
}
